// MLA_59270548685108
// MI455X (gfx1250) — matrix operand order repaired and checked by reading, not compiled, not run on hardware
//
#include <hip/hip_runtime.h>
#include <hip/hip_bf16.h>
#include <stdint.h>

// ---------------- problem constants (from reference) ----------------
constexpr int Bc   = 2;
constexpr int Sc   = 2048;
constexpr int DIMc = 2048;
constexpr int Hc   = 16;
constexpr int NOPEc= 96;
constexpr int ROPEc= 32;
constexpr int HDc  = 128;
constexpr int KVRc = 512;
constexpr int QRc  = 1536;
constexpr int MTOK = Bc * Sc;           // 4096 token rows

// ---------------- types ----------------
typedef __bf16        v16bf __attribute__((ext_vector_type(16)));
typedef float         v8f   __attribute__((ext_vector_type(8)));
typedef unsigned int  u32x4 __attribute__((ext_vector_type(4)));
typedef int           i32x4 __attribute__((ext_vector_type(4)));
typedef int           i32x8 __attribute__((ext_vector_type(8)));

union Frag16 {
    v16bf     v;
    uint32_t  u[8];
    uint4     q[2];
};

__device__ __forceinline__ uint16_t f32_to_bf16(float f) {
    uint32_t x = __builtin_bit_cast(uint32_t, f);
    uint32_t r = x + 0x7fffu + ((x >> 16) & 1u);   // round-to-nearest-even
    return (uint16_t)(r >> 16);
}
__device__ __forceinline__ float bf16_to_f32(uint16_t h) {
    uint32_t x = ((uint32_t)h) << 16;
    return __builtin_bit_cast(float, x);
}

// ---------------- CDNA5 async global->LDS copy (fallback path) -------------
__device__ __forceinline__ void async_ld_b128(uint32_t lds_byte_off,
                                              const uint16_t* g) {
    asm volatile("global_load_async_to_lds_b128 %0, %1, off"
                 :: "v"(lds_byte_off), "v"(g) : "memory");
}
__device__ __forceinline__ void wait_async0() {
#if __has_builtin(__builtin_amdgcn_s_wait_asynccnt)
    __builtin_amdgcn_s_wait_asynccnt(0);
#else
    asm volatile("s_wait_asynccnt 0x0" ::: "memory");
#endif
}

// ---------------- CDNA5 Tensor Data Mover tile load ----------------
// Loads a 128-row x 32-element (bf16) tile, row stride = stride_elems,
// into LDS with hardware padding: 16 DWORDs data + 4 DWORDs pad per row
// -> exactly the LDA=40-element padded LDS layout used by the compute.
#if __has_builtin(__builtin_amdgcn_tensor_load_to_lds)
#define HAVE_TDM 1
__device__ __forceinline__ void tdm_load_tile_128x32(uint32_t lds_off,
                                                     const uint16_t* gaddr,
                                                     int stride_elems) {
    uint64_t ga = (uint64_t)(uintptr_t)gaddr;
    u32x4 g0;
    g0[0] = 1u;                                          // count=1 (valid), user mode
    g0[1] = lds_off;                                     // lds_addr (bytes)
    g0[2] = (uint32_t)ga;                                // global_addr[31:0]
    g0[3] = (uint32_t)((ga >> 32) & 0x01FFFFFFu)         // global_addr[56:32]
          | (2u << 30);                                  // type=2 ("image")
    i32x8 g1;
    g1[0] = (1 << 16)                                    // data_size = 2 bytes
          | (1 << 20)                                    // pad_enable
          | (3 << 22)                                    // pad_interval: 16 DWORDs
          | (3 << 25);                                   // pad_amount:   4 DWORDs
    g1[1] = (int)(32u  << 16);                           // tensor_dim0 = 32
    g1[2] = (int)(128u << 16);                           // tensor_dim1 = 128
    g1[3] = (int)(32u  << 16);                           // tile_dim0 = 32
    g1[4] = 128;                                         // tile_dim1 = 128 (tile_dim2=0)
    g1[5] = stride_elems;                                // tensor_dim0_stride
    g1[6] = 0;
    g1[7] = 0;
    i32x4 z4 = {0, 0, 0, 0};
#if defined(__clang_major__) && (__clang_major__ >= 23)
    i32x8 z8 = {0, 0, 0, 0, 0, 0, 0, 0};
    __builtin_amdgcn_tensor_load_to_lds(g0, g1, z4, z4, z8, 0);
#else
    __builtin_amdgcn_tensor_load_to_lds(g0, g1, z4, z4, 0);
#endif
}
__device__ __forceinline__ void wait_tensor0() {
    __builtin_amdgcn_s_wait_tensorcnt(0);
}
#endif

// ---------------- fp32 -> bf16 (no transpose, for x) ----------------
__global__ void cvt_f32_to_bf16(const float* __restrict__ in,
                                uint16_t* __restrict__ out, int n) {
    int i = blockIdx.x * blockDim.x + threadIdx.x;
    if (i < n) out[i] = f32_to_bf16(in[i]);
}

// ---------------- fp32 [K][N] -> bf16 transposed [N][K] (weights) ----------
__global__ __launch_bounds__(256) void cvt_transpose_kernel(
        const float* __restrict__ in, uint16_t* __restrict__ out, int K, int N) {
    __shared__ float tile[32][33];
    int k0 = blockIdx.y * 32;
    int n0 = blockIdx.x * 32;
    int tx = threadIdx.x;            // 0..31
    int ty = threadIdx.y;            // 0..7
    for (int i = ty; i < 32; i += 8)
        tile[i][tx] = in[(size_t)(k0 + i) * N + n0 + tx];
    __syncthreads();
    for (int i = ty; i < 32; i += 8)
        out[(size_t)(n0 + i) * K + k0 + tx] = f32_to_bf16(tile[tx][i]);
}

// ---------------- bf16 V [B*S][H*HD] -> V^T [B][H*HD][S] ----------------
__global__ __launch_bounds__(256) void transpose_v_kernel(
        const uint16_t* __restrict__ v, uint16_t* __restrict__ vT) {
    __shared__ uint16_t tile[32][34];
    int b  = blockIdx.z;
    int s0 = blockIdx.y * 32;
    int c0 = blockIdx.x * 32;
    int tx = threadIdx.x;
    int ty = threadIdx.y;
    for (int i = ty; i < 32; i += 8)
        tile[i][tx] = v[(size_t)(b * Sc + s0 + i) * (Hc * HDc) + c0 + tx];
    __syncthreads();
    for (int i = ty; i < 32; i += 8)
        vT[(size_t)(b * (Hc * HDc) + c0 + i) * Sc + s0 + tx] = tile[tx][i];
}

// ---------------- bf16 WMMA GEMM, A[M][K] * B^T[N][K] -> C[M][N] ----------
// M,N multiples of 128; K multiple of 32. 256 threads = 8 waves,
// tile 128x128, wave = 32(M) x 64(N). Double-buffered TDM (or async) staging:
// one barrier per K-chunk; DMA of chunk i+1 overlaps WMMA on chunk i.
template <bool OUT_F32>
__global__ __launch_bounds__(256) void gemm_bf16_tn(
        const uint16_t* __restrict__ A, const uint16_t* __restrict__ Bt,
        void* __restrict__ C, int M, int N, int K) {
    constexpr int LDA = 40;                                   // 32 + 8 pad
    __shared__ __align__(16) uint16_t As[2][128 * LDA];
    __shared__ __align__(16) uint16_t Bs[2][128 * LDA];

    const int m0   = blockIdx.y * 128;
    const int n0   = blockIdx.x * 128;
    const int t    = threadIdx.x;
    const int lane = t & 31;
    const int w    = t >> 5;
    const int wm   = (w & 3) * 32;
    const int wn   = (w >> 2) * 64;
    const int l15  = lane & 15;
    const int hi8  = (lane < 16) ? 0 : 8;

    v8f acc[2][4];
    for (int mi = 0; mi < 2; ++mi)
        for (int ni = 0; ni < 4; ++ni)
            for (int e = 0; e < 8; ++e) acc[mi][ni][e] = 0.0f;

    auto issue_tile = [&](int buf, int kc) {
#ifdef HAVE_TDM
        if (w == 0) {
            tdm_load_tile_128x32((uint32_t)(uintptr_t)&As[buf][0],
                                 A  + (size_t)m0 * K + kc, K);
            tdm_load_tile_128x32((uint32_t)(uintptr_t)&Bs[buf][0],
                                 Bt + (size_t)n0 * K + kc, K);
        }
#else
        #pragma unroll
        for (int cc = 0; cc < 2; ++cc) {
            int ch  = t + cc * 256;        // 0..511
            int row = ch >> 2;             // 0..127
            int off = (ch & 3) * 8;        // 0,8,16,24 elements
            async_ld_b128((uint32_t)(uintptr_t)&As[buf][row * LDA + off],
                          A  + (size_t)(m0 + row) * K + kc + off);
            async_ld_b128((uint32_t)(uintptr_t)&Bs[buf][row * LDA + off],
                          Bt + (size_t)(n0 + row) * K + kc + off);
        }
#endif
    };
    auto wait_tiles = [&]() {
#ifdef HAVE_TDM
        if (w == 0) wait_tensor0();
#else
        wait_async0();
#endif
    };

    const int niter = K / 32;
    issue_tile(0, 0);
    for (int i = 0; i < niter; ++i) {
        const int cur = i & 1;
        wait_tiles();
        __syncthreads();            // data ready + everyone done with buf being refilled
        if (i + 1 < niter) issue_tile(cur ^ 1, (i + 1) * 32);

        const uint16_t* Ab = &As[cur][0];
        const uint16_t* Bb = &Bs[cur][0];
        Frag16 af[2], bfr[4];
        {
            const int kb = (lane < 16) ? 0 : 8;    // A: split K halves
            #pragma unroll
            for (int mi = 0; mi < 2; ++mi) {
                const uint16_t* p = Ab + (wm + mi * 16 + l15) * LDA + kb;
                af[mi].q[0] = *(const uint4*)p;
                af[mi].q[1] = *(const uint4*)(p + 16);
            }
        }
        {
            const int kb = (lane < 16) ? 0 : 8;    // B: split K halves
            #pragma unroll
            for (int ni = 0; ni < 4; ++ni) {
                const uint16_t* p = Bb + (wn + ni * 16 + l15) * LDA + kb;
                bfr[ni].q[0] = *(const uint4*)p;
                bfr[ni].q[1] = *(const uint4*)(p + 16);
            }
        }
        #pragma unroll
        for (int mi = 0; mi < 2; ++mi)
            #pragma unroll
            for (int ni = 0; ni < 4; ++ni)
                acc[mi][ni] = __builtin_amdgcn_wmma_f32_16x16x32_bf16(
                    false, af[mi].v, false, bfr[ni].v,
                    (short)0, acc[mi][ni], false, false);
    }

    // ---- epilogue ----
    #pragma unroll
    for (int mi = 0; mi < 2; ++mi)
        #pragma unroll
        for (int ni = 0; ni < 4; ++ni)
            #pragma unroll
            for (int r = 0; r < 8; ++r) {
                int row = m0 + wm + mi * 16 + r + hi8;
                int col = n0 + wn + ni * 16 + l15;
                float v  = acc[mi][ni][r];
                if (OUT_F32)
                    ((float*)C)[(size_t)row * N + col] = v;
                else
                    ((uint16_t*)C)[(size_t)row * N + col] = f32_to_bf16(v);
            }
    (void)M;
}

// ---------------- RoPE on q (in-place, one thread per complex pair) --------
__global__ void rope_q_kernel(uint16_t* __restrict__ q,
                              const float* __restrict__ cosT,
                              const float* __restrict__ sinT) {
    int idx = blockIdx.x * blockDim.x + threadIdx.x;    // MTOK*H*16
    if (idx >= MTOK * Hc * (ROPEc / 2)) return;
    int i   = idx & 15;
    int h   = (idx >> 4) & 15;
    int row = idx >> 8;
    int s   = row & (Sc - 1);
    float c  = cosT[s * (ROPEc / 2) + i];
    float sn = sinT[s * (ROPEc / 2) + i];
    size_t base = (size_t)row * (Hc * HDc) + (size_t)h * HDc + NOPEc + 2 * i;
    float xr = bf16_to_f32(q[base]);
    float xi = bf16_to_f32(q[base + 1]);
    q[base]     = f32_to_bf16(xr * c - xi * sn);
    q[base + 1] = f32_to_bf16(xr * sn + xi * c);
}

// ---------------- assemble k_final: nope copy ----------------
__global__ void copy_nope_kernel(const uint16_t* __restrict__ kn,
                                 uint16_t* __restrict__ kf) {
    int idx = blockIdx.x * blockDim.x + threadIdx.x;    // MTOK*H*96
    if (idx >= MTOK * Hc * NOPEc) return;
    int d   = idx % NOPEc;
    int h   = (idx / NOPEc) & 15;
    int row = idx / (NOPEc * Hc);
    kf[(size_t)row * (Hc * HDc) + (size_t)h * HDc + d] =
        kn[(size_t)row * (Hc * NOPEc) + (size_t)h * NOPEc + d];
}

// ---------------- assemble k_final: roped rope part ----------------
__global__ void rope_k_kernel(const uint16_t* __restrict__ kr,
                              uint16_t* __restrict__ kf,
                              const float* __restrict__ cosT,
                              const float* __restrict__ sinT) {
    int idx = blockIdx.x * blockDim.x + threadIdx.x;    // MTOK*H*16
    if (idx >= MTOK * Hc * (ROPEc / 2)) return;
    int i   = idx & 15;
    int h   = (idx >> 4) & 15;
    int row = idx >> 8;
    int s   = row & (Sc - 1);
    float c  = cosT[s * (ROPEc / 2) + i];
    float sn = sinT[s * (ROPEc / 2) + i];
    size_t src = (size_t)row * (Hc * ROPEc) + (size_t)h * ROPEc + 2 * i;
    float xr = bf16_to_f32(kr[src]);
    float xi = bf16_to_f32(kr[src + 1]);
    size_t dst = (size_t)row * (Hc * HDc) + (size_t)h * HDc + NOPEc + 2 * i;
    kf[dst]     = f32_to_bf16(xr * c - xi * sn);
    kf[dst + 1] = f32_to_bf16(xr * sn + xi * c);
}

// ---------------- flash attention (causal, online softmax) ----------------
// grid = (S/128, H, B), block = 256 (8 independent waves, 16 q-rows each).
// K read row-major from global (contiguous per lane); V read from V^T
// (contiguous per lane). Per-wave causal bound; no block barriers.
__global__ __launch_bounds__(256) void attn_kernel(
        const uint16_t* __restrict__ Q, const uint16_t* __restrict__ Kf,
        const uint16_t* __restrict__ Vt, uint16_t* __restrict__ O) {
    constexpr int PST = 40;
    __shared__ __align__(16) uint16_t Pl[8][16 * PST];   // per-wave P bounce

    const int b = blockIdx.z, h = blockIdx.y, qb = blockIdx.x;
    const int t = threadIdx.x, lane = t & 31, w = t >> 5;
    const int q0  = qb * 128 + w * 16;
    const int hi8 = (lane < 16) ? 0 : 8;
    const int l15 = lane & 15;
    const size_t rstr = (size_t)Hc * HDc;    // 2048
    const uint16_t* qp  = Q  + (size_t)(b * Sc) * rstr + (size_t)h * HDc;
    const uint16_t* kp  = Kf + (size_t)(b * Sc) * rstr + (size_t)h * HDc;
    const uint16_t* vtp = Vt + ((size_t)b * (Hc * HDc) + (size_t)h * HDc) * Sc;

    // Q fragments (16 rows x 128 dims, 4 K-chunks of 32)
    Frag16 qf[4];
    #pragma unroll
    for (int c = 0; c < 4; ++c) {
        int kb = c * 32 + ((lane < 16) ? 0 : 8);
        const uint16_t* p = qp + (size_t)(q0 + l15) * rstr + kb;
        qf[c].q[0] = *(const uint4*)p;
        qf[c].q[1] = *(const uint4*)(p + 16);
    }

    float mrow[8], lrow[8];
    v8f   oacc[8];
    #pragma unroll
    for (int r = 0; r < 8; ++r) { mrow[r] = -1e30f; lrow[r] = 0.0f; }
    #pragma unroll
    for (int ni = 0; ni < 8; ++ni)
        for (int e = 0; e < 8; ++e) oacc[ni][e] = 0.0f;

    const float scale = 0.08838834764831845f;   // 1/sqrt(128)
    const int ntiles = (q0 >> 5) + 1;           // per-wave causal bound

    for (int jt = 0; jt < ntiles; ++jt) {
        const int j0 = jt * 32;
        if (jt + 1 < ntiles)                     // prefetch next K tile
            __builtin_prefetch(kp + (size_t)(j0 + 32 + l15) * rstr, 0, 0);

        // --- S = Q K^T for two 16-key sub-tiles ---
        v8f sacc[2];
        #pragma unroll
        for (int nt2 = 0; nt2 < 2; ++nt2) {
            for (int e = 0; e < 8; ++e) sacc[nt2][e] = 0.0f;
            int key = j0 + nt2 * 16 + l15;
            #pragma unroll
            for (int c = 0; c < 4; ++c) {
                int kd = c * 32 + ((lane < 16) ? 0 : 8);
                Frag16 kfrag;
                const uint16_t* p = kp + (size_t)key * rstr + kd;
                kfrag.q[0] = *(const uint4*)p;
                kfrag.q[1] = *(const uint4*)(p + 16);
                sacc[nt2] = __builtin_amdgcn_wmma_f32_16x16x32_bf16(
                    false, qf[c].v, false, kfrag.v,
                    (short)0, sacc[nt2], false, false);
            }
        }
        // scale + causal mask
        #pragma unroll
        for (int nt2 = 0; nt2 < 2; ++nt2) {
            int key = j0 + nt2 * 16 + l15;
            #pragma unroll
            for (int r = 0; r < 8; ++r) {
                int qrow = q0 + r + hi8;
                float sv = sacc[nt2][r] * scale;
                sacc[nt2][r] = (key <= qrow) ? sv : -1e30f;
            }
        }
        // row max across 16-lane group; online-softmax update
        float newm[8];
        #pragma unroll
        for (int r = 0; r < 8; ++r) {
            float mx = fmaxf(sacc[0][r], sacc[1][r]);
            #pragma unroll
            for (int off = 8; off >= 1; off >>= 1)
                mx = fmaxf(mx, __shfl_xor(mx, off, 32));
            newm[r] = fmaxf(mrow[r], mx);
        }
        // P = exp(s - m_new); stash bf16 P in per-wave LDS for A-layout
        #pragma unroll
        for (int nt2 = 0; nt2 < 2; ++nt2)
            #pragma unroll
            for (int r = 0; r < 8; ++r) {
                float pv = __expf(sacc[nt2][r] - newm[r]);
                sacc[nt2][r] = pv;
                Pl[w][(r + hi8) * PST + nt2 * 16 + l15] = f32_to_bf16(pv);
            }
        #pragma unroll
        for (int r = 0; r < 8; ++r) {
            float ps = sacc[0][r] + sacc[1][r];
            #pragma unroll
            for (int off = 8; off >= 1; off >>= 1)
                ps += __shfl_xor(ps, off, 32);
            float alpha = __expf(mrow[r] - newm[r]);
            lrow[r] = lrow[r] * alpha + ps;
            mrow[r] = newm[r];
            #pragma unroll
            for (int ni = 0; ni < 8; ++ni) oacc[ni][r] *= alpha;
        }

        // --- O += P * V  (V fragments straight from V^T, contiguous) ---
        Frag16 pf;
        {
            int kb = (lane < 16) ? 0 : 8;
            const uint16_t* p = &Pl[w][l15 * PST + kb];
            pf.q[0] = *(const uint4*)p;
            pf.q[1] = *(const uint4*)(p + 16);
        }
        #pragma unroll
        for (int ni = 0; ni < 8; ++ni) {
            Frag16 vfrag;
            int kb = (lane < 16) ? 0 : 8;
            const uint16_t* p = vtp + (size_t)(ni * 16 + l15) * Sc + j0 + kb;
            vfrag.q[0] = *(const uint4*)p;
            vfrag.q[1] = *(const uint4*)(p + 16);
            oacc[ni] = __builtin_amdgcn_wmma_f32_16x16x32_bf16(
                false, pf.v, false, vfrag.v,
                (short)0, oacc[ni], false, false);
        }
    }

    // normalize + write attn_out (bf16)
    #pragma unroll
    for (int r = 0; r < 8; ++r) {
        float inv = (lrow[r] > 0.0f) ? (1.0f / lrow[r]) : 0.0f;
        int row = q0 + r + hi8;
        #pragma unroll
        for (int ni = 0; ni < 8; ++ni) {
            size_t off = (size_t)(b * Sc + row) * rstr
                       + (size_t)h * HDc + ni * 16 + l15;
            O[off] = f32_to_bf16(oacc[ni][r] * inv);
        }
    }
}

// ---------------- host orchestration ----------------
extern "C" void kernel_launch(void* const* d_in, const int* in_sizes, int n_in,
                              void* d_out, int out_size, void* d_ws, size_t ws_size,
                              hipStream_t stream) {
    (void)in_sizes; (void)n_in; (void)out_size; (void)ws_size;
    const float* x    = (const float*)d_in[0];
    const float* cosT = (const float*)d_in[1];
    const float* sinT = (const float*)d_in[2];
    const float* wqd  = (const float*)d_in[3];
    const float* wqu  = (const float*)d_in[4];
    const float* wkvd = (const float*)d_in[5];
    const float* wupn = (const float*)d_in[6];
    const float* wupr = (const float*)d_in[7];
    const float* wupv = (const float*)d_in[8];
    const float* wo   = (const float*)d_in[9];

    uint16_t* p = (uint16_t*)d_ws;
    auto alloc = [&](size_t n) { uint16_t* r = p; p += n; return r; };
    uint16_t* xb     = alloc((size_t)MTOK * DIMc);          // x bf16 (row-major)
    uint16_t* wqd_t  = alloc((size_t)DIMc * QRc);           // transposed weights
    uint16_t* wqu_t  = alloc((size_t)QRc * Hc * HDc);
    uint16_t* wkvd_t = alloc((size_t)DIMc * KVRc);
    uint16_t* wupn_t = alloc((size_t)KVRc * Hc * NOPEc);
    uint16_t* wupr_t = alloc((size_t)KVRc * Hc * ROPEc);
    uint16_t* wupv_t = alloc((size_t)KVRc * Hc * HDc);
    uint16_t* wo_t   = alloc((size_t)Hc * HDc * DIMc);
    uint16_t* q_lat  = alloc((size_t)MTOK * QRc);
    uint16_t* qbuf   = alloc((size_t)MTOK * Hc * HDc);
    uint16_t* kv_lat = alloc((size_t)MTOK * KVRc);
    uint16_t* k_nope = alloc((size_t)MTOK * Hc * NOPEc);
    uint16_t* k_rope = alloc((size_t)MTOK * Hc * ROPEc);
    uint16_t* kfin   = alloc((size_t)MTOK * Hc * HDc);
    uint16_t* vbuf   = alloc((size_t)MTOK * Hc * HDc);
    uint16_t* vT     = alloc((size_t)Bc * Hc * HDc * Sc);
    uint16_t* aout   = alloc((size_t)MTOK * Hc * HDc);

    // x -> bf16
    {
        size_t n = (size_t)MTOK * DIMc;
        cvt_f32_to_bf16<<<dim3((unsigned)((n + 255) / 256)), 256, 0, stream>>>(
            x, xb, (int)n);
    }
    // weights -> bf16, transposed [N][K]
    auto cvtT = [&](const float* in, uint16_t* out, int K, int N) {
        cvt_transpose_kernel<<<dim3(N / 32, K / 32), dim3(32, 8), 0, stream>>>(
            in, out, K, N);
    };
    cvtT(wqd,  wqd_t,  DIMc, QRc);
    cvtT(wqu,  wqu_t,  QRc,  Hc * HDc);
    cvtT(wkvd, wkvd_t, DIMc, KVRc);
    cvtT(wupn, wupn_t, KVRc, Hc * NOPEc);
    cvtT(wupr, wupr_t, KVRc, Hc * ROPEc);
    cvtT(wupv, wupv_t, KVRc, Hc * HDc);
    cvtT(wo,   wo_t,   Hc * HDc, DIMc);

    auto gemm = [&](const uint16_t* A, const uint16_t* Bt, void* C,
                    int M, int N, int K, bool outF32) {
        dim3 g((unsigned)(N / 128), (unsigned)(M / 128));
        if (outF32) gemm_bf16_tn<true ><<<g, 256, 0, stream>>>(A, Bt, C, M, N, K);
        else        gemm_bf16_tn<false><<<g, 256, 0, stream>>>(A, Bt, C, M, N, K);
    };

    // projections
    gemm(xb,     wqd_t,  q_lat,  MTOK, QRc,        DIMc, false); // q latent
    gemm(q_lat,  wqu_t,  qbuf,   MTOK, Hc * HDc,   QRc,  false); // q
    gemm(xb,     wkvd_t, kv_lat, MTOK, KVRc,       DIMc, false); // kv latent
    gemm(kv_lat, wupn_t, k_nope, MTOK, Hc * NOPEc, KVRc, false); // k nope
    gemm(kv_lat, wupr_t, k_rope, MTOK, Hc * ROPEc, KVRc, false); // k rope
    gemm(kv_lat, wupv_t, vbuf,   MTOK, Hc * HDc,   KVRc, false); // v

    // rope + k assembly
    {
        int n1 = MTOK * Hc * (ROPEc / 2);
        rope_q_kernel<<<dim3((n1 + 255) / 256), 256, 0, stream>>>(qbuf, cosT, sinT);
        int n2 = MTOK * Hc * NOPEc;
        copy_nope_kernel<<<dim3((n2 + 255) / 256), 256, 0, stream>>>(k_nope, kfin);
        rope_k_kernel<<<dim3((n1 + 255) / 256), 256, 0, stream>>>(k_rope, kfin, cosT, sinT);
    }

    // V -> V^T for contiguous PV fragments
    transpose_v_kernel<<<dim3((Hc * HDc) / 32, Sc / 32, Bc), dim3(32, 8), 0, stream>>>(
        vbuf, vT);

    // flash attention
    attn_kernel<<<dim3(Sc / 128, Hc, Bc), 256, 0, stream>>>(qbuf, kfin, vT, aout);

    // output projection -> f32 d_out
    gemm(aout, wo_t, d_out, MTOK, DIMc, Hc * HDc, true);
}
